// DeepWuKongModel_807453851674
// MI455X (gfx1250) — matrix operand order repaired and checked by reading, not compiled, not run on hardware
//
#include <hip/hip_runtime.h>
#include <hip/hip_bf16.h>
#include <stdint.h>

#define HID 128
#define NLAYERS 3

typedef __attribute__((ext_vector_type(16))) __bf16          v16bf;
typedef __attribute__((ext_vector_type(8)))  float           v8f;
typedef __attribute__((ext_vector_type(4)))  unsigned short  us4;
typedef __attribute__((ext_vector_type(8)))  unsigned short  us8;
typedef __attribute__((ext_vector_type(16))) unsigned short  us16;
typedef __attribute__((ext_vector_type(4)))  float           f4;

__device__ __forceinline__ unsigned short f2bf_bits(float f) {
  // round-to-nearest-even float32 -> bf16 bits
  unsigned u = __float_as_uint(f);
  return (unsigned short)((u + 0x7FFFu + ((u >> 16) & 1u)) >> 16);
}

// ---------------- utility kernels ----------------

__global__ void fill_f32(float* __restrict__ p, float v, size_t n) {
  size_t i = (size_t)blockIdx.x * blockDim.x + threadIdx.x;
  if (i < n) p[i] = v;
}

// W[l][k][n] (f32) -> Wt[l][n][k] (bf16): per-lane-contiguous K for the WMMA B fragment
__global__ void cvt_bf16_T(const float* __restrict__ W, unsigned short* __restrict__ Wt, int total) {
  int i = blockIdx.x * blockDim.x + threadIdx.x;
  if (i >= total) return;
  int l = i >> 14;            // / (128*128)
  int k = (i >> 7) & 127;
  int n = i & 127;
  Wt[((size_t)l << 14) + n * HID + k] = f2bf_bits(W[i]);
}

// ---------------- GCN normalization ----------------

__global__ void deg_count(const int* __restrict__ dst, float* __restrict__ deg, int E) {
  int e = blockIdx.x * blockDim.x + threadIdx.x;
  if (e < E) atomicAdd(&deg[dst[e]], 1.0f);
}

__global__ void deg_to_dis(float* __restrict__ deg, int N) {
  int i = blockIdx.x * blockDim.x + threadIdx.x;
  if (i < N) deg[i] = rsqrtf(deg[i]);   // deg >= 1 (self loop), matches reference
}

__global__ void edge_norm(const int* __restrict__ src, const int* __restrict__ dst,
                          const float* __restrict__ dis, float* __restrict__ nrm, int E) {
  int e = blockIdx.x * blockDim.x + threadIdx.x;
  if (e < E) nrm[e] = dis[src[e]] * dis[dst[e]];
}

// ---- WMMA GEMM + fused epilogue:
//   hW  = relu?(H) @ W           (bf16 WMMA, f32 accumulate)
//   agg = bias + hW * dis^2      (self-loop + bias pre-seeded for the edge scatter)
// One 256-thread block (8 waves) per 32-row panel; wave w owns output cols [16w,16w+16)
// and applies its register-resident B fragments to TWO 16-row A tiles (8 WMMAs/wave).

__global__ void __launch_bounds__(256)
gcn_gemm(const float* __restrict__ H, const unsigned short* __restrict__ Wt,
         const float* __restrict__ dis, const float* __restrict__ bias,
         float* __restrict__ hW, float* __restrict__ agg,
         int nRows, int preRelu) {
  // stride 136 elems = 272 B: rows 16B-aligned AND 4-bank staggered (conflict-free)
  __shared__ __align__(16) unsigned short lA[32][136];
  const int tid = threadIdx.x;
  const int m0  = blockIdx.x << 5;

  // stage 32x128 A panel, f32 -> bf16, optional fused ReLU (1024 float4 groups)
  for (int i = tid; i < 32 * (HID / 4); i += 256) {
    int r = i >> 5, c4 = (i & 31) << 2;
    int row = m0 + r; if (row >= nRows) row = nRows - 1;
    f4 v = *(const f4*)(H + (size_t)row * HID + c4);
    if (preRelu) {
      v[0] = fmaxf(v[0], 0.0f); v[1] = fmaxf(v[1], 0.0f);
      v[2] = fmaxf(v[2], 0.0f); v[3] = fmaxf(v[3], 0.0f);
    }
    us4 o;
    o[0] = f2bf_bits(v[0]); o[1] = f2bf_bits(v[1]);
    o[2] = f2bf_bits(v[2]); o[3] = f2bf_bits(v[3]);
    *(us4*)&lA[r][c4] = o;
  }
  __syncthreads();

  const int lane = tid & 31;
  const int n0   = (tid >> 5) << 4;
  const int hi   = lane >> 4;          // half-wave select
  const int arow = lane & 15;          // A: row = lane%16
  const int ah   = hi << 3;            // A: lanes16-31 hold K offsets +8
  const int bh   = hi << 3;            // B: lanes16-31 hold K offsets +8
  const int col  = lane & 15;          // B/C/D column
  const unsigned short* wrow = Wt + (size_t)(n0 + col) * HID;

  // preload ALL B fragments for this wave's 16 columns (4 K-chunks, 32 VGPRs)
  us16 bu[4];
  #pragma unroll
  for (int k = 0; k < 4; ++k) {
    us8 blo = *(const us8*)(wrow + k * 32 + bh);
    us8 bhi = *(const us8*)(wrow + k * 32 + 16 + bh);
    for (int i = 0; i < 8; ++i) { bu[k][i] = blo[i]; bu[k][i + 8] = bhi[i]; }
  }

  v8f acc0 = {}, acc1 = {};
  #pragma unroll
  for (int k = 0; k < 4; ++k) {
    const int kk = k * 32;
    // A frag: lane<16 -> K {kk+0..7, kk+16..23}; lane>=16 -> +8. Two 16B LDS loads/tile.
    us8 a0lo = *(const us8*)&lA[arow][kk + ah];
    us8 a0hi = *(const us8*)&lA[arow][kk + 16 + ah];
    us8 a1lo = *(const us8*)&lA[16 + arow][kk + ah];
    us8 a1hi = *(const us8*)&lA[16 + arow][kk + 16 + ah];
    us16 a0, a1;
    #pragma unroll
    for (int i = 0; i < 8; ++i) {
      a0[i] = a0lo[i]; a0[i + 8] = a0hi[i];
      a1[i] = a1lo[i]; a1[i + 8] = a1hi[i];
    }
    acc0 = __builtin_amdgcn_wmma_f32_16x16x32_bf16(false, __builtin_bit_cast(v16bf, a0),
                                                   false, __builtin_bit_cast(v16bf, bu[k]),
                                                   (short)0, acc0, false, false);
    acc1 = __builtin_amdgcn_wmma_f32_16x16x32_bf16(false, __builtin_bit_cast(v16bf, a1),
                                                   false, __builtin_bit_cast(v16bf, bu[k]),
                                                   (short)0, acc1, false, false);
  }

  // C/D 16x16 f32 layout: VGPR r -> row r (lanes 0-15) / row r+8 (lanes 16-31)
  const float bb  = bias[n0 + col];
  const int   rb0 = m0 + (hi << 3);
  const int   rb1 = rb0 + 16;
  if (m0 + 32 <= nRows) {                      // fast path: full panel, no guards
    #pragma unroll
    for (int r = 0; r < 8; ++r) {
      int row0 = rb0 + r, row1 = rb1 + r;
      size_t o0 = (size_t)row0 * HID + n0 + col;
      size_t o1 = (size_t)row1 * HID + n0 + col;
      float d0 = dis[row0], d1 = dis[row1];
      hW[o0]  = acc0[r];
      hW[o1]  = acc1[r];
      agg[o0] = bb + acc0[r] * d0 * d0;        // bias + self-loop seed
      agg[o1] = bb + acc1[r] * d1 * d1;
    }
  } else {                                     // tail panel
    #pragma unroll
    for (int r = 0; r < 8; ++r) {
      int row0 = rb0 + r, row1 = rb1 + r;
      if (row0 < nRows) {
        size_t o = (size_t)row0 * HID + n0 + col;
        float d = dis[row0];
        hW[o] = acc0[r]; agg[o] = bb + acc0[r] * d * d;
      }
      if (row1 < nRows) {
        size_t o = (size_t)row1 * HID + n0 + col;
        float d = dis[row1];
        hW[o] = acc1[r]; agg[o] = bb + acc1[r] * d * d;
      }
    }
  }
}

// ---------------- message passing ----------------

// 32 threads per edge, float4 per thread: gather hW[src], scale, atomic scatter to dst
__global__ void __launch_bounds__(256)
edge_scatter(const float* __restrict__ hW, const int* __restrict__ src,
             const int* __restrict__ dst, const float* __restrict__ nrm,
             float* __restrict__ out, int E) {
  int e = blockIdx.x * 8 + (threadIdx.x >> 5);
  if (e >= E) return;
  int c4 = (threadIdx.x & 31) << 2;
  int s = src[e], d = dst[e];
  float w = nrm[e];
  f4 v = *(const f4*)(hW + (size_t)s * HID + c4);
  float* o = out + (size_t)d * HID + c4;
  atomicAdd(o + 0, v[0] * w);
  atomicAdd(o + 1, v[1] * w);
  atomicAdd(o + 2, v[2] * w);
  atomicAdd(o + 3, v[3] * w);
}

// ---------------- pooling + head ----------------

// 32 threads per node, float4 per thread; final-layer ReLU fused on load.
__global__ void __launch_bounds__(256)
pool_k(const float* __restrict__ h, const int* __restrict__ batch,
       float* __restrict__ psum, unsigned* __restrict__ pmax,
       float* __restrict__ cnt, int N) {
  int n = blockIdx.x * 8 + (threadIdx.x >> 5);
  if (n >= N) return;
  int c4 = (threadIdx.x & 31) << 2;
  int g = batch[n];
  f4 v = *(const f4*)(h + (size_t)n * HID + c4);
  #pragma unroll
  for (int j = 0; j < 4; ++j) {
    float vj = fmaxf(v[j], 0.0f);
    atomicAdd(&psum[g * HID + c4 + j], vj);
    // vj >= 0, so uint ordering of the bit pattern == float ordering
    atomicMax(&pmax[g * HID + c4 + j], __float_as_uint(vj));
  }
  if ((threadIdx.x & 31) == 0) atomicAdd(&cnt[g], 1.0f);
}

__global__ void __launch_bounds__(128)
ffn_k(const float* __restrict__ psum, const unsigned* __restrict__ pmax,
      const float* __restrict__ cnt, const float* __restrict__ Wffn,
      const float* __restrict__ bffn, float* __restrict__ hidden) {
  int g = blockIdx.x, j = threadIdx.x;
  float invc = 1.0f / fmaxf(cnt[g], 1.0f);
  float acc = bffn[j];
  for (int k = 0; k < HID; ++k) {
    float mx = __uint_as_float(pmax[g * HID + k]);      // 0 for empty graphs
    float mn = psum[g * HID + k] * invc;
    acc += mx * Wffn[k * HID + j] + mn * Wffn[(HID + k) * HID + j];
  }
  hidden[g * HID + j] = fmaxf(acc, 0.0f);
}

__global__ void final_k(const float* __restrict__ hidden, const float* __restrict__ Wfin,
                        const float* __restrict__ bfin, float* __restrict__ out, int B) {
  int i = blockIdx.x * blockDim.x + threadIdx.x;
  if (i >= B * 2) return;
  int g = i >> 1, c = i & 1;
  float acc = bfin[c];
  for (int j = 0; j < HID; ++j) acc += hidden[g * HID + j] * Wfin[j * 2 + c];
  out[i] = acc;
}

// ---------------- launch ----------------

extern "C" void kernel_launch(void* const* d_in, const int* in_sizes, int n_in,
                              void* d_out, int out_size, void* d_ws, size_t ws_size,
                              hipStream_t stream) {
  const float* x    = (const float*)d_in[0];
  const float* Wc   = (const float*)d_in[1];
  const float* bc   = (const float*)d_in[2];
  const float* Wffn = (const float*)d_in[3];
  const float* bffn = (const float*)d_in[4];
  const float* Wfin = (const float*)d_in[5];
  const float* bfin = (const float*)d_in[6];
  const int*   eidx = (const int*)d_in[7];
  const int*   batc = (const int*)d_in[8];

  const int E = in_sizes[7] / 2;
  const int N = in_sizes[8];
  const int B = out_size / 2;
  const int* esrc = eidx;
  const int* edst = eidx + E;

  char* w = (char*)d_ws;
  auto carve = [&](size_t bytes) -> void* {
    void* p = (void*)w;
    w += (bytes + 255) & ~(size_t)255;
    return p;
  };
  float*          bufA  = (float*)carve((size_t)N * HID * 4);          // h @ Wc
  float*          bufB  = (float*)carve((size_t)N * HID * 4);          // aggregated h
  float*          dis   = (float*)carve((size_t)N * 4);                // deg -> D^-1/2
  float*          nrmE  = (float*)carve((size_t)E * 4);                // per-edge norm
  unsigned short* wcb   = (unsigned short*)carve((size_t)NLAYERS * HID * HID * 2);
  float*          psum  = (float*)carve((size_t)B * HID * 4);
  unsigned*       pmax  = (unsigned*)carve((size_t)B * HID * 4);
  float*          cntb  = (float*)carve((size_t)B * 4);
  float*          hidden= (float*)carve((size_t)B * HID * 4);

  const unsigned bN = (unsigned)((N + 255) / 256);
  const unsigned bE = (unsigned)((E + 255) / 256);

  // normalization: deg (with self-loop) -> rsqrt -> per-edge norm
  fill_f32<<<bN, 256, 0, stream>>>(dis, 1.0f, (size_t)N);
  deg_count<<<bE, 256, 0, stream>>>(edst, dis, E);
  deg_to_dis<<<bN, 256, 0, stream>>>(dis, N);
  edge_norm<<<bE, 256, 0, stream>>>(esrc, edst, dis, nrmE, E);

  // bf16 transposed weights for the WMMA B fragments
  const int wtot = NLAYERS * HID * HID;
  cvt_bf16_T<<<(wtot + 255) / 256, 256, 0, stream>>>(Wc, wcb, wtot);

  const float* cur = x;
  for (int l = 0; l < NLAYERS; ++l) {
    gcn_gemm<<<(N + 31) / 32, 256, 0, stream>>>(cur, wcb + ((size_t)l << 14),
                                                dis, bc + (size_t)l * HID,
                                                bufA, bufB, N, l > 0);
    edge_scatter<<<(E + 7) / 8, 256, 0, stream>>>(bufA, esrc, edst, nrmE, bufB, E);
    cur = bufB;
  }

  // pooling (zero-init the accumulators each call — workspace is not re-poisoned)
  fill_f32<<<(B * HID + 255) / 256, 256, 0, stream>>>(psum, 0.0f, (size_t)B * HID);
  fill_f32<<<(B * HID + 255) / 256, 256, 0, stream>>>((float*)pmax, 0.0f, (size_t)B * HID);
  fill_f32<<<(B + 255) / 256, 256, 0, stream>>>(cntb, 0.0f, (size_t)B);
  pool_k<<<(N + 7) / 8, 256, 0, stream>>>(cur, batc, psum, pmax, cntb, N);

  // head
  ffn_k<<<B, 128, 0, stream>>>(psum, pmax, cntb, Wffn, bffn, hidden);
  final_k<<<(B * 2 + 255) / 256, 256, 0, stream>>>(hidden, Wfin, bfin, (float*)d_out, B);
}
